// GraphConv_41987600286095
// MI455X (gfx1250) — matrix operand order repaired and checked by reading, not compiled, not run on hardware
//
#include <hip/hip_runtime.h>

#define NB 8
#define NN 2048
#define DF 128

typedef __attribute__((ext_vector_type(16))) __bf16 v16bf;
typedef __attribute__((ext_vector_type(8)))  __bf16 v8bf;
typedef __attribute__((ext_vector_type(8)))  float  v8f;

static __device__ __forceinline__ v16bf cat16(v8bf lo, v8bf hi) {
    return __builtin_shufflevector(lo, hi, 0,1,2,3,4,5,6,7,8,9,10,11,12,13,14,15);
}

// ---------------------------------------------------------------------------
// Kernel 0: zero the column-sum accumulator (ws is poisoned by the harness).
// ---------------------------------------------------------------------------
__global__ void k_zero(float* __restrict__ p) {
    p[blockIdx.x * 256 + threadIdx.x] = 0.0f;
}

// ---------------------------------------------------------------------------
// Kernel 1: stream A once: copy A -> d_out (first output of the tuple) and
// accumulate column sums d[b][j] = sum_i A[b][i][j] via f32 global atomics.
// One block = 32 rows of one batch; float4 loads/stores, 8 col-partials/thread.
// ---------------------------------------------------------------------------
__global__ __launch_bounds__(256) void k_copy_colsum(const float* __restrict__ A,
                                                     float* __restrict__ outA,
                                                     float* __restrict__ colsum) {
    const int b    = blockIdx.y;
    const int row0 = blockIdx.x * 32;
    const int tid  = threadIdx.x;

    float part[8];
#pragma unroll
    for (int i = 0; i < 8; ++i) part[i] = 0.0f;

    const size_t base = ((size_t)b * NN + row0) * NN;
    const float4* src = reinterpret_cast<const float4*>(A + base);
    float4*       dst = reinterpret_cast<float4*>(outA + base);

    for (int r = 0; r < 32; ++r) {
        const int rb = r * (NN / 4);
#pragma unroll
        for (int g = 0; g < 2; ++g) {
            const int i4 = g * 256 + tid;
            float4 v = src[rb + i4];
            dst[rb + i4] = v;
            part[g * 4 + 0] += v.x;
            part[g * 4 + 1] += v.y;
            part[g * 4 + 2] += v.z;
            part[g * 4 + 3] += v.w;
        }
    }
#pragma unroll
    for (int g = 0; g < 2; ++g) {
        const int col = (g * 256 + tid) * 4;
#pragma unroll
        for (int c2 = 0; c2 < 4; ++c2)
            atomicAdd(&colsum[b * NN + col + c2], part[g * 4 + c2]);
    }
}

// ---------------------------------------------------------------------------
// Kernel 2a: rs[b][j] = (colsum + 1)^-1/2   (the +1 is the diagonal of A+I)
// ---------------------------------------------------------------------------
__global__ void k_rs(const float* __restrict__ colsum, float* __restrict__ rs) {
    const int i = blockIdx.x * 256 + threadIdx.x;
    rs[i] = rsqrtf(colsum[i] + 1.0f);
}

// ---------------------------------------------------------------------------
// Kernel 2b: XsT[b][f][j] = bf16( X[b][j][f] * rs[b][j] )  (transposed + scaled
// so a WMMA B-fragment is two contiguous 16B runs per lane).
// ---------------------------------------------------------------------------
__global__ void k_xst(const float* __restrict__ X, const float* __restrict__ rs,
                      __bf16* __restrict__ XsT) {
    const int idx = blockIdx.x * 256 + threadIdx.x;   // < NB*DF*NN
    const int j   = idx % NN;
    const int bf  = idx / NN;
    const int f   = bf % DF;
    const int b   = bf / DF;
    const float v = X[((size_t)b * NN + j) * DF + f] * rs[b * NN + j];
    XsT[idx] = (__bf16)v;
}

// ---------------------------------------------------------------------------
// Kernel 2c: Wpk[fo][fi] = bf16( W[fi][fo] )  (W transposed for GEMM2's
// A-fragments: Hᵀ = Wᵀ ⊗ DADHᵀ).
// ---------------------------------------------------------------------------
__global__ void k_wpk(const float* __restrict__ W, __bf16* __restrict__ Wpk) {
    const int idx = blockIdx.x * 256 + threadIdx.x;   // < DF*DF
    const int fi  = idx % DF;
    const int fo  = idx / DF;
    Wpk[idx] = (__bf16)W[fi * DF + fo];
}

// ---------------------------------------------------------------------------
// Kernel 3: main fused kernel. Block = 64 threads (2 waves), one 16-row tile.
// Wave w accumulates K in [w*1024,(w+1)*1024) with v_wmma_f32_16x16x32_bf16,
// cross-wave reduce through LDS, epilogue adds the +I term and d_i scaling,
// then both waves run GEMM2 (Hᵀ tiles) from an LDS-staged bf16 DADH tile.
// ---------------------------------------------------------------------------
__global__ __launch_bounds__(64) void k_gcn(const float* __restrict__ A,
                                            const float* __restrict__ X,
                                            const float* __restrict__ rs,
                                            const __bf16* __restrict__ XsT,
                                            const __bf16* __restrict__ Wpk,
                                            float* __restrict__ H) {
    __shared__ float  red[16 * DF];        // f32 partial accumulators (wave 1)
    __shared__ __bf16 dadh[16 * 136];      // bf16 DADH tile, padded stride

    const int b    = blockIdx.y;
    const int row0 = blockIdx.x * 16;
    const int tid  = threadIdx.x;
    const int wave = tid >> 5;
    const int lane = tid & 31;
    const int c    = lane & 15;            // A row / B col / C col within tile
    const int h    = lane >> 4;            // lane half

    v8f acc[8];
#pragma unroll
    for (int f = 0; f < 8; ++f)
#pragma unroll
        for (int e = 0; e < 8; ++e) acc[f][e] = 0.0f;

    const int kstart = wave * (NN / 2);
    const int kend   = kstart + (NN / 2);
    const float*  arow  = A + ((size_t)b * NN + row0 + c) * NN;
    const __bf16* xst_b = XsT + (size_t)b * DF * NN;

    for (int kb = kstart; kb < kend; kb += 32) {
        // --- A fragment: rows of A, f32 -> bf16 in registers.
        // 16-bit A layout: element e <-> K = kb + 16*(e/8) + 8*h + (e%8)
        const float* ap = arow + kb + 8 * h;
        float4 a0 = *reinterpret_cast<const float4*>(ap);
        float4 a1 = *reinterpret_cast<const float4*>(ap + 4);
        float4 a2 = *reinterpret_cast<const float4*>(ap + 16);
        float4 a3 = *reinterpret_cast<const float4*>(ap + 20);
        v16bf af;
        af[0]  = (__bf16)a0.x; af[1]  = (__bf16)a0.y; af[2]  = (__bf16)a0.z; af[3]  = (__bf16)a0.w;
        af[4]  = (__bf16)a1.x; af[5]  = (__bf16)a1.y; af[6]  = (__bf16)a1.z; af[7]  = (__bf16)a1.w;
        af[8]  = (__bf16)a2.x; af[9]  = (__bf16)a2.y; af[10] = (__bf16)a2.z; af[11] = (__bf16)a2.w;
        af[12] = (__bf16)a3.x; af[13] = (__bf16)a3.y; af[14] = (__bf16)a3.z; af[15] = (__bf16)a3.w;

#pragma unroll
        for (int f = 0; f < 8; ++f) {
            // B layout: element e <-> K = kb + 16*(e/8) + 8*h + (e%8), column = f*16 + c
            const __bf16* bp = xst_b + ((size_t)(f * 16 + c)) * NN + kb + 8 * h;
            v8bf blo = *reinterpret_cast<const v8bf*>(bp);
            v8bf bhi = *reinterpret_cast<const v8bf*>(bp + 16);
            v16bf bfrag = cat16(blo, bhi);
            acc[f] = __builtin_amdgcn_wmma_f32_16x16x32_bf16(
                false, af, false, bfrag, (short)0, acc[f], false, false);
        }
    }

    // --- cross-wave K reduction through LDS (C layout: M = e+8h, N = c)
    if (wave == 1) {
#pragma unroll
        for (int f = 0; f < 8; ++f)
#pragma unroll
            for (int e = 0; e < 8; ++e)
                red[(e + 8 * h) * DF + f * 16 + c] = acc[f][e];
    }
    __syncthreads();

    if (wave == 0) {
        float rr[8];
#pragma unroll
        for (int e = 0; e < 8; ++e) rr[e] = rs[b * NN + row0 + e + 8 * h];
#pragma unroll
        for (int f = 0; f < 8; ++f) {
#pragma unroll
            for (int e = 0; e < 8; ++e) {
                const int rl   = e + 8 * h;
                const int feat = f * 16 + c;
                float v = acc[f][e] + red[rl * DF + feat];
                // DADH = d_i * (sum_j A*Xs + d_i*X[i,:])  (Xs already has d_j)
                v = (v + X[((size_t)b * NN + row0 + rl) * DF + feat] * rr[e]) * rr[e];
                dadh[rl * 136 + feat] = (__bf16)v;
            }
        }
    }
    __syncthreads();

    // --- GEMM2: Hᵀ[fo][i] = sum_fi Wᵀ[fo][fi] * DADHᵀ[fi][i], relu, store.
    v16bf bfr[4];
#pragma unroll
    for (int q = 0; q < 4; ++q) {
        const __bf16* dp = &dadh[c * 136 + q * 32 + 8 * h];
        bfr[q] = cat16(*reinterpret_cast<const v8bf*>(dp),
                       *reinterpret_cast<const v8bf*>(dp + 16));
    }
#pragma unroll
    for (int t = 0; t < 4; ++t) {
        const int fo_t = wave * 4 + t;     // waves split the 8 fo tiles
        v8f acc2;
#pragma unroll
        for (int e = 0; e < 8; ++e) acc2[e] = 0.0f;
#pragma unroll
        for (int q = 0; q < 4; ++q) {
            const __bf16* wp = Wpk + (fo_t * 16 + c) * DF + q * 32 + 8 * h;
            v16bf wf = cat16(*reinterpret_cast<const v8bf*>(wp),
                             *reinterpret_cast<const v8bf*>(wp + 16));
            acc2 = __builtin_amdgcn_wmma_f32_16x16x32_bf16(
                false, wf, false, bfr[q], (short)0, acc2, false, false);
        }
        // D element e: fo = fo_t*16 + 8h + e (contiguous), i = row0 + c
        float* hp = H + ((size_t)b * NN + row0 + c) * DF + fo_t * 16 + 8 * h;
        float4 o0, o1;
        o0.x = fmaxf(acc2[0], 0.0f); o0.y = fmaxf(acc2[1], 0.0f);
        o0.z = fmaxf(acc2[2], 0.0f); o0.w = fmaxf(acc2[3], 0.0f);
        o1.x = fmaxf(acc2[4], 0.0f); o1.y = fmaxf(acc2[5], 0.0f);
        o1.z = fmaxf(acc2[6], 0.0f); o1.w = fmaxf(acc2[7], 0.0f);
        *reinterpret_cast<float4*>(hp)     = o0;
        *(reinterpret_cast<float4*>(hp) + 1) = o1;
    }
}

// ---------------------------------------------------------------------------
extern "C" void kernel_launch(void* const* d_in, const int* in_sizes, int n_in,
                              void* d_out, int out_size, void* d_ws, size_t ws_size,
                              hipStream_t stream) {
    const float* A = (const float*)d_in[0];   // [8,2048,2048] f32
    const float* X = (const float*)d_in[1];   // [8,2048,128]  f32
    const float* W = (const float*)d_in[2];   // [128,128]     f32

    float* outA = (float*)d_out;                                // A passthrough
    float* outH = (float*)d_out + (size_t)NB * NN * NN;         // H

    // workspace layout (~4.2 MB total)
    float*  colsum = (float*)d_ws;                              // 16384 f32
    float*  rs     = colsum + NB * NN;                          // 16384 f32
    __bf16* XsT    = (__bf16*)(rs + NB * NN);                   // 2M bf16
    __bf16* Wpk    = XsT + (size_t)NB * DF * NN;                // 16K bf16

    k_zero       <<<dim3(NB * NN / 256), 256, 0, stream>>>(colsum);
    k_copy_colsum<<<dim3(NN / 32, NB),   256, 0, stream>>>(A, outA, colsum);
    k_rs         <<<dim3(NB * NN / 256), 256, 0, stream>>>(colsum, rs);
    k_xst        <<<dim3(NB * DF * NN / 256), 256, 0, stream>>>(X, rs, XsT);
    k_wpk        <<<dim3(DF * DF / 256), 256, 0, stream>>>(W, Wpk);
    k_gcn        <<<dim3(NN / 16, NB), 64, 0, stream>>>(A, X, rs, XsT, Wpk, outH);
}
